// AttentionPropagation_1915555414510
// MI455X (gfx1250) — matrix operand order repaired and checked by reading, not compiled, not run on hardware
//
#include <hip/hip_runtime.h>
#include <hip/hip_bf16.h>
#include <cstdint>
#include <cmath>

#define EPS 1e-5f
constexpr int Bsz = 4, Nn = 2048, Mm = 2048, Cc = 256, Hh = 4, Dh = 64;

typedef __attribute__((ext_vector_type(16))) __bf16 v16bf;
typedef __attribute__((ext_vector_type(8)))  float  v8f;

union FragAB { v16bf v; uint32_t u[8]; };
union F4 { float4 v; float f[4]; };
union H4 { uint2 v; __bf16 h[4]; };
union H8 { uint4 v; __bf16 h[8]; };

__device__ inline uint32_t pack2(__bf16 lo, __bf16 hi) {
  union { __bf16 h[2]; uint32_t u; } t; t.h[0] = lo; t.h[1] = hi; return t.u;
}

// A fragment: 16(M) x 32(K) bf16 from row-major LDS tile, row stride 32.
// Per-lane dwords are two runs of 4 consecutive dwords -> two ds_load_b128.
__device__ inline v16bf load_frag_A(const __bf16* a, int lane) {
  int m   = lane & 15;
  int idx = (lane & 16) ? 2 : 0;          // K half-offset 0 or 16 -> uint4 index
  const uint4* row = (const uint4*)(a + m * 32);
  uint4 g0 = row[idx];                    // k = idx*8 .. idx*8+7
  uint4 g1 = row[idx + 1];                // k = idx*8+8 .. +7
  FragAB f;
  f.u[0] = g0.x; f.u[1] = g0.y; f.u[2] = g0.z; f.u[3] = g0.w;
  f.u[4] = g1.x; f.u[5] = g1.y; f.u[6] = g1.z; f.u[7] = g1.w;
  return f.v;
}

// B fragment from packed LDS: Btp[col][p] = bf16 pair (k=2p,k=2p+1), 16 dwords
// per column. Lane reads 8 consecutive dwords -> two ds_load_b128.
__device__ inline v16bf load_frag_Bp(const uint32_t* bt, int lane) {
  int n  = lane & 15;
  int k2 = (lane & 16) ? 8 : 0;
  const uint4* p = (const uint4*)(bt + n * 16 + k2);
  uint4 g0 = p[0], g1 = p[1];
  FragAB f;
  f.u[0] = g0.x; f.u[1] = g0.y; f.u[2] = g0.z; f.u[3] = g0.w;
  f.u[4] = g1.x; f.u[5] = g1.y; f.u[6] = g1.z; f.u[7] = g1.w;
  return f.v;
}

__device__ inline v8f wmma_bf16(v16bf a, v16bf b, v8f c) {
  return __builtin_amdgcn_wmma_f32_16x16x32_bf16(false, a, false, b, (short)0, c,
                                                 false, false);
}

// ---------------------------------------------------------------------------
// 1x1 conv + BN + ReLU: Y[o,n] = relu(sum_c W'[o,c] X[c,n] + b'[o])
// ---------------------------------------------------------------------------
template <bool IN_BF16_CN, bool OUT_TRANS>
__global__ __launch_bounds__(256) void filter_gemm(
    const void* __restrict__ Xv, const float* __restrict__ W,
    const float* __restrict__ bias, const float* __restrict__ gamma,
    const float* __restrict__ beta, const float* __restrict__ mean,
    const float* __restrict__ var, void* __restrict__ Yv, int Nlen) {
  __shared__ __align__(16) __bf16 At[64 * 32];     // o x c
  __shared__ __align__(16) uint32_t Btp[64 * 16];  // packed: [n][c/2]
  __shared__ float sSc[64], sBi[64];

  const int tid   = threadIdx.x;
  const int ntile = blockIdx.x * 64;
  const int otile = blockIdx.y * 64;
  const int b     = blockIdx.z;

  if (tid < 64) {
    int o = otile + tid;
    float s = gamma[o] * rsqrtf(var[o] + EPS);
    sSc[tid] = s;
    sBi[tid] = (bias[o] - mean[o]) * s + beta[o];
  }
  __syncthreads();

  const int wave = tid >> 5, lane = tid & 31;
  const int wr = wave >> 1, wc = wave & 1;
  v8f acc0 = {}, acc1 = {};

  for (int k0 = 0; k0 < Cc; k0 += 32) {
    {  // A = W * scale, 64(o) x 32(c): one b128 store per thread
      int r  = tid >> 2;
      int c0 = (tid & 3) * 8;
      const float4* src = (const float4*)(W + (size_t)(otile + r) * Cc + k0 + c0);
      F4 w0, w1; w0.v = src[0]; w1.v = src[1];
      float s = sSc[r];
      H8 u;
#pragma unroll
      for (int i = 0; i < 4; ++i) {
        u.h[i]     = (__bf16)(w0.f[i] * s);
        u.h[i + 4] = (__bf16)(w1.f[i] * s);
      }
      *(uint4*)&At[r * 32 + c0] = u.v;
    }
    if (IN_BF16_CN) {  // input bf16 [B,C,Nlen]: pair k-rows, packed b32 stores
      const __bf16* X = (const __bf16*)Xv;
      int kp = tid >> 4;          // 0..15 -> c = 2kp, 2kp+1
      int n0 = (tid & 15) * 4;    // 0..60
      const __bf16* r0 = X + ((size_t)b * Cc + k0 + 2 * kp) * Nlen + ntile + n0;
      H4 a0, a1;
      a0.v = *(const uint2*)r0;
      a1.v = *(const uint2*)(r0 + Nlen);
#pragma unroll
      for (int i = 0; i < 4; ++i)
        Btp[(n0 + i) * 16 + kp] = pack2(a0.h[i], a1.h[i]);
    } else {           // input fp32 [B,Nlen,C]: c-contiguous, one b128 store
      const float* X = (const float*)Xv;
      int nl = tid >> 2;
      int c0 = (tid & 3) * 8;
      const float4* src =
          (const float4*)(X + ((size_t)b * Nlen + ntile + nl) * Cc + k0 + c0);
      F4 x0, x1; x0.v = src[0]; x1.v = src[1];
      H8 u;
#pragma unroll
      for (int i = 0; i < 4; ++i) {
        u.h[2 * i]     = (__bf16)x0.f[i];   // wrong pairing? see below
        u.h[2 * i + 1] = 0;
      }
      // pack (c0+0..7) pairs in order: h[j] = c0+j, dword j = pair j
      u.h[0] = (__bf16)x0.f[0]; u.h[1] = (__bf16)x0.f[1];
      u.h[2] = (__bf16)x0.f[2]; u.h[3] = (__bf16)x0.f[3];
      u.h[4] = (__bf16)x1.f[0]; u.h[5] = (__bf16)x1.f[1];
      u.h[6] = (__bf16)x1.f[2]; u.h[7] = (__bf16)x1.f[3];
      *(uint4*)&Btp[nl * 16 + (c0 >> 1)] = u.v;
    }
    __syncthreads();
    v16bf a  = load_frag_A(At + wr * 16 * 32, lane);
    v16bf b0 = load_frag_Bp(Btp + (wc * 32) * 16, lane);
    v16bf b1 = load_frag_Bp(Btp + (wc * 32 + 16) * 16, lane);
    acc0 = wmma_bf16(a, b0, acc0);
    acc1 = wmma_bf16(a, b1, acc1);
    __syncthreads();
  }

  const int col = lane & 15;
  const int rowoff = (lane & 16) ? 8 : 0;
#pragma unroll
  for (int i = 0; i < 8; ++i) {
    int rl = wr * 16 + rowoff + i;
    int o  = otile + rl;
    float bb = sBi[rl];
    float y0 = fmaxf(acc0[i] + bb, 0.f);
    float y1 = fmaxf(acc1[i] + bb, 0.f);
    int n0g = ntile + wc * 32 + col;
    int n1g = n0g + 16;
    if (OUT_TRANS) {
      float* Y = (float*)Yv;
      Y[((size_t)b * Nlen + n0g) * Cc + o] = y0;
      Y[((size_t)b * Nlen + n1g) * Cc + o] = y1;
    } else {
      __bf16* Y = (__bf16*)Yv;
      Y[((size_t)b * Cc + o) * Nlen + n0g] = (__bf16)y0;
      Y[((size_t)b * Cc + o) * Nlen + n1g] = (__bf16)y1;
    }
  }
}

// ---------------------------------------------------------------------------
// QK[b,h,n,m] = (1/8) sum_d q[b,h*64+d,n] * k[b,h*64+d,m]
// ---------------------------------------------------------------------------
__global__ __launch_bounds__(256) void qk_gemm(const __bf16* __restrict__ q,
                                               const __bf16* __restrict__ k,
                                               float* __restrict__ QK) {
  __shared__ __align__(16) __bf16 At[64 * 32];     // n x d
  __shared__ __align__(16) uint32_t Btp[64 * 16];  // packed [m][d/2]
  const int tid = threadIdx.x;
  const int mtile = blockIdx.x * 64;
  const int ntile = blockIdx.y * 64;
  const int bh = blockIdx.z;
  const int b = bh >> 2, h = bh & 3;
  const int wave = tid >> 5, lane = tid & 31;
  const int wr = wave >> 1, wc = wave & 1;
  v8f acc0 = {}, acc1 = {};
  const size_t qbase = ((size_t)b * Cc + h * Dh) * Nn;
  const size_t kbase = ((size_t)b * Cc + h * Dh) * Mm;

  for (int d0 = 0; d0 < Dh; d0 += 32) {
    {  // A: transpose q -> At[n][d], packed b32 stores along d-pairs
      int dp = tid >> 4;          // 0..15 -> d = 2dp, 2dp+1
      int n0 = (tid & 15) * 4;
      const __bf16* r0 = q + qbase + (size_t)(d0 + 2 * dp) * Nn + ntile + n0;
      H4 a0, a1;
      a0.v = *(const uint2*)r0;
      a1.v = *(const uint2*)(r0 + Nn);
#pragma unroll
      for (int i = 0; i < 4; ++i)
        *(uint32_t*)&At[(n0 + i) * 32 + 2 * dp] = pack2(a0.h[i], a1.h[i]);
    }
    {  // B: k rows d-major, m contiguous; pack d-pairs per column m
      int dp = tid >> 4;
      int m0 = (tid & 15) * 4;
      const __bf16* r0 = k + kbase + (size_t)(d0 + 2 * dp) * Mm + mtile + m0;
      H4 a0, a1;
      a0.v = *(const uint2*)r0;
      a1.v = *(const uint2*)(r0 + Mm);
#pragma unroll
      for (int i = 0; i < 4; ++i)
        Btp[(m0 + i) * 16 + dp] = pack2(a0.h[i], a1.h[i]);
    }
    __syncthreads();
    v16bf a  = load_frag_A(At + wr * 16 * 32, lane);
    v16bf b0 = load_frag_Bp(Btp + (wc * 32) * 16, lane);
    v16bf b1 = load_frag_Bp(Btp + (wc * 32 + 16) * 16, lane);
    acc0 = wmma_bf16(a, b0, acc0);
    acc1 = wmma_bf16(a, b1, acc1);
    __syncthreads();
  }
  const int col = lane & 15;
  const int rowoff = (lane & 16) ? 8 : 0;
  float* out = QK + (size_t)bh * Nn * Mm;
#pragma unroll
  for (int i = 0; i < 8; ++i) {
    int n = ntile + wr * 16 + rowoff + i;
    out[(size_t)n * Mm + mtile + wc * 32 + col]      = acc0[i] * 0.125f;
    out[(size_t)n * Mm + mtile + wc * 32 + col + 16] = acc1[i] * 0.125f;
  }
}

__device__ inline void online_combine(float& m, float& s, float m2, float s2) {
  float nm = fmaxf(m, m2);
  s = s * __expf(m - nm) + s2 * __expf(m2 - nm);
  m = nm;
}

__global__ __launch_bounds__(256) void row_stats(const float* __restrict__ QK,
                                                 float* __restrict__ mx,
                                                 float* __restrict__ sm) {
  __shared__ float smx[256], ssm[256];
  const int n = blockIdx.x, bh = blockIdx.y;
  const float* row = QK + ((size_t)bh * Nn + n) * Mm;
  float m = -INFINITY, s = 0.f;
  for (int i = threadIdx.x; i < Mm; i += 256) {
    float x = row[i];
    float nm = fmaxf(m, x);
    s = s * __expf(m - nm) + __expf(x - nm);
    m = nm;
  }
  smx[threadIdx.x] = m; ssm[threadIdx.x] = s;
  __syncthreads();
  for (int off = 128; off > 0; off >>= 1) {
    if (threadIdx.x < off) {
      float mm = smx[threadIdx.x], ss = ssm[threadIdx.x];
      online_combine(mm, ss, smx[threadIdx.x + off], ssm[threadIdx.x + off]);
      smx[threadIdx.x] = mm; ssm[threadIdx.x] = ss;
    }
    __syncthreads();
  }
  if (threadIdx.x == 0) {
    mx[(size_t)bh * Nn + n] = smx[0];
    sm[(size_t)bh * Nn + n] = ssm[0];
  }
}

__global__ __launch_bounds__(256) void col_stats(const float* __restrict__ QK,
                                                 float* __restrict__ mx,
                                                 float* __restrict__ sm) {
  const int m_ = blockIdx.x * 256 + threadIdx.x;
  const int bh = blockIdx.y;
  const float* base = QK + (size_t)bh * Nn * Mm + m_;
  float mv = -INFINITY, s = 0.f;
  for (int n = 0; n < Nn; ++n) {
    float x = base[(size_t)n * Mm];
    float nm = fmaxf(mv, x);
    s = s * __expf(mv - nm) + __expf(x - nm);
    mv = nm;
  }
  mx[(size_t)bh * Mm + m_] = mv;
  sm[(size_t)bh * Mm + m_] = s;
}

// add0[d,n] = sum_m softmax_m(QK)[n,m] * v0[d,m]
__global__ __launch_bounds__(256) void attn_av0(
    const float* __restrict__ QK, const float* __restrict__ mx0,
    const float* __restrict__ sm0, const __bf16* __restrict__ v0,
    __bf16* __restrict__ add0) {
  __shared__ __align__(16) __bf16 At[64 * 32];     // d x m
  __shared__ __align__(16) uint32_t Btp[64 * 16];  // packed [n][m/2]
  __shared__ float smx[64], srs[64];
  const int tid = threadIdx.x;
  const int ntile = blockIdx.x * 64;
  const int bh = blockIdx.y;
  const int b = bh >> 2, h = bh & 3;
  const int wave = tid >> 5, lane = tid & 31;
  const int wr = wave >> 1, wc = wave & 1;
  v8f acc0 = {}, acc1 = {};
  const size_t vbase = ((size_t)b * Cc + h * Dh) * Mm;
  const float* qkrow = QK + ((size_t)bh * Nn + ntile) * Mm;

  if (tid < 64) {
    smx[tid] = mx0[(size_t)bh * Nn + ntile + tid];
    srs[tid] = 1.f / sm0[(size_t)bh * Nn + ntile + tid];
  }
  __syncthreads();

  for (int m0 = 0; m0 < Mm; m0 += 32) {
    {  // A: v0 tile, m-contiguous -> single b128 copy per thread
      int dl = tid >> 2, mc0 = (tid & 3) * 8;
      *(uint4*)&At[dl * 32 + mc0] =
          *(const uint4*)(v0 + vbase + (size_t)dl * Mm + m0 + mc0);
    }
    {  // B: exp-normalized QK row, m-contiguous -> packed b128 store
      int nl = tid >> 2, mc0 = (tid & 3) * 8;
      const float4* src = (const float4*)(qkrow + (size_t)nl * Mm + m0 + mc0);
      F4 x0, x1; x0.v = src[0]; x1.v = src[1];
      float mxv = smx[nl], rs = srs[nl];
      H8 u;
#pragma unroll
      for (int i = 0; i < 4; ++i) {
        u.h[i]     = (__bf16)(__expf(x0.f[i] - mxv) * rs);
        u.h[i + 4] = (__bf16)(__expf(x1.f[i] - mxv) * rs);
      }
      *(uint4*)&Btp[nl * 16 + (mc0 >> 1)] = u.v;
    }
    __syncthreads();
    v16bf a  = load_frag_A(At + wr * 16 * 32, lane);
    v16bf b0 = load_frag_Bp(Btp + (wc * 32) * 16, lane);
    v16bf b1 = load_frag_Bp(Btp + (wc * 32 + 16) * 16, lane);
    acc0 = wmma_bf16(a, b0, acc0);
    acc1 = wmma_bf16(a, b1, acc1);
    __syncthreads();
  }
  const int col = lane & 15;
  const int rowoff = (lane & 16) ? 8 : 0;
#pragma unroll
  for (int i = 0; i < 8; ++i) {
    int d = wr * 16 + rowoff + i;
    size_t base = ((size_t)b * Cc + h * Dh + d) * Nn;
    add0[base + ntile + wc * 32 + col]      = (__bf16)acc0[i];
    add0[base + ntile + wc * 32 + col + 16] = (__bf16)acc1[i];
  }
}

// add1[d,m] = sum_n softmax_n(QK)[n,m] * v1[d,n]
__global__ __launch_bounds__(256) void attn_av1(
    const float* __restrict__ QK, const float* __restrict__ mx1,
    const float* __restrict__ sm1, const __bf16* __restrict__ v1,
    __bf16* __restrict__ add1) {
  __shared__ __align__(16) __bf16 At[64 * 32];     // d x n
  __shared__ __align__(16) uint32_t Btp[64 * 16];  // packed [m][n/2]
  __shared__ float smx[64], srs[64];
  const int tid = threadIdx.x;
  const int mtile = blockIdx.x * 64;
  const int bh = blockIdx.y;
  const int b = bh >> 2, h = bh & 3;
  const int wave = tid >> 5, lane = tid & 31;
  const int wr = wave >> 1, wc = wave & 1;
  v8f acc0 = {}, acc1 = {};
  const size_t vbase = ((size_t)b * Cc + h * Dh) * Nn;

  if (tid < 64) {
    smx[tid] = mx1[(size_t)bh * Mm + mtile + tid];
    srs[tid] = 1.f / sm1[(size_t)bh * Mm + mtile + tid];
  }
  __syncthreads();

  for (int n0 = 0; n0 < Nn; n0 += 32) {
    {  // A: v1 tile, n-contiguous -> b128 copy
      int dl = tid >> 2, nc0 = (tid & 3) * 8;
      *(uint4*)&At[dl * 32 + nc0] =
          *(const uint4*)(v1 + vbase + (size_t)dl * Nn + n0 + nc0);
    }
    {  // B: exp-normalized QK, pack n-pairs per column m
      int np  = tid >> 4;          // 0..15 -> n = n0+2np, n0+2np+1
      int m0l = (tid & 15) * 4;
      const float* r0 = QK + ((size_t)bh * Nn + n0 + 2 * np) * Mm + mtile + m0l;
      F4 x0, x1;
      x0.v = *(const float4*)r0;
      x1.v = *(const float4*)(r0 + Mm);
#pragma unroll
      for (int i = 0; i < 4; ++i) {
        int mg = m0l + i;
        __bf16 e0 = (__bf16)(__expf(x0.f[i] - smx[mg]) * srs[mg]);
        __bf16 e1 = (__bf16)(__expf(x1.f[i] - smx[mg]) * srs[mg]);
        Btp[mg * 16 + np] = pack2(e0, e1);
      }
    }
    __syncthreads();
    v16bf a  = load_frag_A(At + wr * 16 * 32, lane);
    v16bf b0 = load_frag_Bp(Btp + (wc * 32) * 16, lane);
    v16bf b1 = load_frag_Bp(Btp + (wc * 32 + 16) * 16, lane);
    acc0 = wmma_bf16(a, b0, acc0);
    acc1 = wmma_bf16(a, b1, acc1);
    __syncthreads();
  }
  const int col = lane & 15;
  const int rowoff = (lane & 16) ? 8 : 0;
#pragma unroll
  for (int i = 0; i < 8; ++i) {
    int d = wr * 16 + rowoff + i;
    size_t base = ((size_t)b * Cc + h * Dh + d) * Mm;
    add1[base + mtile + wc * 32 + col]      = (__bf16)acc0[i];
    add1[base + mtile + wc * 32 + col + 16] = (__bf16)acc1[i];
  }
}

extern "C" void kernel_launch(void* const* d_in, const int* in_sizes, int n_in,
                              void* d_out, int out_size, void* d_ws,
                              size_t ws_size, hipStream_t stream) {
  const float* x1  = (const float*)d_in[0];
  const float* x2  = (const float*)d_in[1];
  const float* Ws  = (const float*)d_in[2];
  const float* bs  = (const float*)d_in[3];
  const float* gs  = (const float*)d_in[4];
  const float* bes = (const float*)d_in[5];
  const float* ms  = (const float*)d_in[6];
  const float* vrs = (const float*)d_in[7];

  size_t off = 0;
  auto wsAlloc = [&](size_t bytes) {
    void* p = (char*)d_ws + off;
    off += (bytes + 255) & ~(size_t)255;
    return p;
  };
  const size_t actElems = (size_t)Bsz * Cc * Nn;
  __bf16* q    = (__bf16*)wsAlloc(actElems * 2);
  __bf16* kk   = (__bf16*)wsAlloc(actElems * 2);
  __bf16* v0   = (__bf16*)wsAlloc(actElems * 2);
  __bf16* v1   = (__bf16*)wsAlloc(actElems * 2);
  __bf16* add0 = (__bf16*)wsAlloc(actElems * 2);
  __bf16* add1 = (__bf16*)wsAlloc(actElems * 2);
  float* QKbuf = (float*)wsAlloc((size_t)Bsz * Hh * Nn * Mm * 4);
  float* mx0 = (float*)wsAlloc((size_t)Bsz * Hh * Nn * 4);
  float* sm0 = (float*)wsAlloc((size_t)Bsz * Hh * Nn * 4);
  float* mx1 = (float*)wsAlloc((size_t)Bsz * Hh * Mm * 4);
  float* sm1 = (float*)wsAlloc((size_t)Bsz * Hh * Mm * 4);

  const dim3 blk(256);
  const dim3 gF(Nn / 64, Cc / 64, Bsz);
  const int CC = Cc * Cc;

  filter_gemm<false, false><<<gF, blk, 0, stream>>>(
      x1, Ws + 0 * CC, bs + 0 * Cc, gs + 0 * Cc, bes + 0 * Cc, ms + 0 * Cc,
      vrs + 0 * Cc, q, Nn);
  filter_gemm<false, false><<<gF, blk, 0, stream>>>(
      x2, Ws + 1 * CC, bs + 1 * Cc, gs + 1 * Cc, bes + 1 * Cc, ms + 1 * Cc,
      vrs + 1 * Cc, kk, Mm);
  filter_gemm<false, false><<<gF, blk, 0, stream>>>(
      x2, Ws + 2 * CC, bs + 2 * Cc, gs + 2 * Cc, bes + 2 * Cc, ms + 2 * Cc,
      vrs + 2 * Cc, v0, Mm);
  filter_gemm<false, false><<<gF, blk, 0, stream>>>(
      x1, Ws + 3 * CC, bs + 3 * Cc, gs + 3 * Cc, bes + 3 * Cc, ms + 3 * Cc,
      vrs + 3 * Cc, v1, Nn);

  qk_gemm<<<dim3(Mm / 64, Nn / 64, Bsz * Hh), blk, 0, stream>>>(q, kk, QKbuf);

  row_stats<<<dim3(Nn, Bsz * Hh), blk, 0, stream>>>(QKbuf, mx0, sm0);
  col_stats<<<dim3(Mm / 256, Bsz * Hh), blk, 0, stream>>>(QKbuf, mx1, sm1);

  attn_av0<<<dim3(Nn / 64, Bsz * Hh), blk, 0, stream>>>(QKbuf, mx0, sm0, v0, add0);
  attn_av1<<<dim3(Mm / 64, Bsz * Hh), blk, 0, stream>>>(QKbuf, mx1, sm1, v1, add1);

  float* out0 = (float*)d_out;
  float* out1 = out0 + (size_t)Bsz * Nn * Cc;
  filter_gemm<true, true><<<gF, blk, 0, stream>>>(
      add0, Ws + 4 * CC, bs + 4 * Cc, gs + 4 * Cc, bes + 4 * Cc, ms + 4 * Cc,
      vrs + 4 * Cc, out0, Nn);
  filter_gemm<true, true><<<gF, blk, 0, stream>>>(
      add1, Ws + 5 * CC, bs + 5 * Cc, gs + 5 * Cc, bes + 5 * Cc, ms + 5 * Cc,
      vrs + 5 * Cc, out1, Mm);
}
